// ExpertLayer_82480551952471
// MI455X (gfx1250) — matrix operand order repaired and checked by reading, not compiled, not run on hardware
//
#include <hip/hip_runtime.h>

// ---------------------------------------------------------------------------
// MoE expert layer for MI455X (gfx1250): bf16 WMMA GEMMs with f32 accumulate.
// D=1024, E=8, H=2048, tokens = B*S = 8192.
// ---------------------------------------------------------------------------

typedef __attribute__((ext_vector_type(8)))  float        v8f;
typedef __attribute__((ext_vector_type(16))) __bf16       v16bf;
typedef __attribute__((ext_vector_type(8)))  unsigned int v8u;
typedef __attribute__((ext_vector_type(4)))  int          v4i;

// Pointer types for the async-DMA builtin: global (as1) src, LDS (as3) dst.
typedef __attribute__((address_space(1))) v4i gl_v4i;
typedef __attribute__((address_space(3))) v4i lds_v4i;

static constexpr int TOK = 8192;   // B*S
static constexpr int DIM = 1024;
static constexpr int HID = 2048;
static constexpr int NE  = 8;

#if defined(__HIP_DEVICE_COMPILE__) && \
    __has_builtin(__builtin_amdgcn_global_load_async_to_lds_b128)
#define HAS_ASYNC_LDS 1
#else
#define HAS_ASYNC_LDS 0
#endif

__device__ __forceinline__ void wait_async0() {
#if __has_builtin(__builtin_amdgcn_s_wait_asynccnt)
    __builtin_amdgcn_s_wait_asynccnt(0);
#else
    asm volatile("s_wait_asynccnt 0x0" ::: "memory");
#endif
}

__device__ __forceinline__ v8f wmma_bf16(v8u a, v8u b, v8f c) {
    return __builtin_amdgcn_wmma_f32_16x16x32_bf16(
        /*neg_a=*/false, __builtin_bit_cast(v16bf, a),
        /*neg_b=*/false, __builtin_bit_cast(v16bf, b),
        /*c_mod=*/(short)0, c, /*reuse_a=*/false, /*reuse_b=*/false);
}

__device__ __forceinline__ unsigned int pack_bf16_pair(float lo, float hi) {
    unsigned int ulo = (unsigned int)__builtin_bit_cast(unsigned short, (__bf16)lo);
    unsigned int uhi = (unsigned int)__builtin_bit_cast(unsigned short, (__bf16)hi);
    return ulo | (uhi << 16);
}

// ---------------------------------------------------------------------------
// Pack a row-major f32 weight matrix W[K][N] into bf16 WMMA B-fragment tiles.
// Tile (32k x 16n): 32 lanes x 8 dwords; lane L, dword j holds
// K = kt*32 + (L>=16 ? 8 : 0) + (j<4 ? 2j : 16 + 2(j-4)) (+1 in high half), N = nt*16 + (L&15).
// Tile order: nt-major (the K-stream of one N-tile is contiguous).
// ---------------------------------------------------------------------------
__global__ __launch_bounds__(256) void pack_b_kernel(
    const float* __restrict__ W, unsigned int* __restrict__ dst,
    int N, int kshift, int total) {
    int d = blockIdx.x * 256 + threadIdx.x;
    if (d >= total) return;
    int j    = d & 7;
    int lane = (d >> 3) & 31;
    int tile = d >> 8;
    int ktG  = tile & ((1 << kshift) - 1);
    int ntG  = tile >> kshift;
    int k = ktG * 32 + ((lane >> 4) << 3) + ((j < 4) ? 2 * j : 16 + 2 * (j - 4));
    int n = ntG * 16 + (lane & 15);
    float lo = W[(size_t)k * N + n];
    float hi = W[(size_t)(k + 1) * N + n];
    dst[d] = pack_bf16_pair(lo, hi);
}

// ---------------------------------------------------------------------------
// GEMM: C[M,N] = epilogue(A[M,K] @ B[K,N] + bias);  K, N compile-time.
//   AT   : float (cvt->bf16 while staging) or __bf16 (direct copy / async DMA)
//   MODE : 0 = bias        -> bf16  (input projection h)
//          1 = bias+relu   -> bf16  (expert layer 1)
//          2 = ew[t,e]*relu(bias+acc) (+=) -> f32 (expert layer 2 combine)
//          3 = bias        -> f32  (output projection, pre-LN)
// Block tile 64x128, 8 waves; wave w: m-tile = w&3, n-tiles = (w>>2)*4..+3.
// A is staged per 64-wide K chunk into LDS in A-fragment order (async DMA on
// the bf16 path); B fragments are register double-buffered so loads for kt+1
// fly under the WMMAs of kt; both A fragments load right after the barrier.
// ---------------------------------------------------------------------------
template <typename AT, int MODE, int K, int N>
__global__ __launch_bounds__(256) void gemm_bf16_kernel(
    const AT* __restrict__ A, const unsigned int* __restrict__ Bp,
    const float* __restrict__ bias,
    float* __restrict__ outF, __bf16* __restrict__ outB,
    const float* __restrict__ ew,
    int expert, int accumulate) {

    constexpr int BM = 64;
    constexpr int K32 = K / 32;
    constexpr int TSTR = K32 * 256;   // dwords between n-tiles in packed B

    const int tid  = threadIdx.x;
    const int lane = tid & 31;
    const int wave = tid >> 5;
    const int mBlock = blockIdx.x * BM;
    const int nBlock = blockIdx.y * 128;

    __shared__ unsigned int ldsA[2][4][32][8];  // [ktl][m-tile][lane][dword]
    __shared__ float ldsEW[BM];

    if (MODE == 2 && tid < BM)
        ldsEW[tid] = ew[(size_t)(mBlock + tid) * NE + expert];

    const int mt     = wave & 3;
    const int ntBase = (wave >> 2) * 4;

    // Per-lane base into packed B for this wave's first n-tile.
    const unsigned int* Bl =
        Bp + (size_t)((nBlock >> 4) + ntBase) * TSTR + lane * 8;

    v8f acc[4];
#pragma unroll
    for (int t = 0; t < 4; ++t)
#pragma unroll
        for (int i = 0; i < 8; ++i) acc[t][i] = 0.0f;

    v8u bcur[4], bnxt[4];
#pragma unroll
    for (int t = 0; t < 4; ++t)
        bcur[t] = *(const v8u*)(Bl + (size_t)t * TSTR);
#pragma unroll
    for (int t = 0; t < 4; ++t) bnxt[t] = bcur[t];

#pragma unroll 1
    for (int kc = 0; kc < K; kc += 64) {
        if (kc) __syncthreads();  // previous chunk's LDS reads done

        // ---- stage A chunk (64 rows x 64 K) in fragment order ----
        if constexpr (sizeof(AT) == 2 && HAS_ASYNC_LDS) {
#if HAS_ASYNC_LDS
            // Each 16B chunk of a fragment is 16B-contiguous in row-major
            // bf16 A -> pure DMA copy, no VGPR round trip.
#pragma unroll
            for (int i = 0; i < 2; ++i) {
                int c    = tid * 2 + i;          // 512 chunks of 16B
                int half = c & 1;
                int ln   = (c >> 1) & 31;
                int smt  = (c >> 6) & 3;
                int ktl  = (c >> 8) & 1;
                int kb   = (ln >> 4) * 8;
                int ks   = ktl * 32 + (half ? 16 + kb : kb);
                int gm   = mBlock + smt * 16 + (ln & 15);
                const AT* src = &A[(size_t)gm * K + kc + ks];
                unsigned int* dst = &ldsA[ktl][smt][ln][half * 4];
                __builtin_amdgcn_global_load_async_to_lds_b128(
                    (gl_v4i*)src, (lds_v4i*)dst, 0, 0);
            }
            wait_async0();
#endif
        } else {
#pragma unroll
            for (int i = 0; i < 8; ++i) {
                int d   = tid * 8 + i;          // 2048 dwords total
                int j   = d & 7;
                int ln  = (d >> 3) & 31;
                int smt = (d >> 8) & 3;
                int ktl = d >> 10;
                int kb  = (ln >> 4) * 8;
                int kk  = (j < 4) ? (kb + 2 * j) : (16 + kb + 2 * (j - 4));
                int gm  = mBlock + smt * 16 + (ln & 15);
                int gk  = kc + ktl * 32 + kk;
                unsigned int w;
                if constexpr (sizeof(AT) == 2) {
                    w = *(const unsigned int*)(&A[(size_t)gm * K + gk]);
                } else {
                    float2 f = *(const float2*)(&A[(size_t)gm * K + gk]);
                    w = pack_bf16_pair(f.x, f.y);
                    if (i == 0 && kc + 64 < K)
                        __builtin_prefetch(&A[(size_t)gm * K + gk + 64], 0, 3);
                }
                ldsA[ktl][smt][ln][j] = w;
            }
        }
        __syncthreads();

        // ---- compute: 2 K-steps of 32; B double-buffered in registers;
        //      both A fragments fetched immediately after the barrier ----
        v8u a0 = *(const v8u*)(&ldsA[0][mt][lane][0]);
        v8u a1 = *(const v8u*)(&ldsA[1][mt][lane][0]);
#pragma unroll
        for (int ktl = 0; ktl < 2; ++ktl) {
            const int kt = (kc >> 5) + ktl;
            if (kt + 1 < K32) {
#pragma unroll
                for (int t = 0; t < 4; ++t)
                    bnxt[t] = *(const v8u*)(Bl + (size_t)(kt + 1) * 256 +
                                            (size_t)t * TSTR);
            }
            v8u a = (ktl == 0) ? a0 : a1;
#pragma unroll
            for (int t = 0; t < 4; ++t) acc[t] = wmma_bf16(a, bcur[t], acc[t]);
#pragma unroll
            for (int t = 0; t < 4; ++t) bcur[t] = bnxt[t];
        }
    }

    // ---- epilogue (D layout: lanes 0-15 -> M=v, lanes 16-31 -> M=8+v) ----
    const int rowBase = (lane >> 4) * 8;
#pragma unroll
    for (int t = 0; t < 4; ++t) {
        int n = nBlock + (ntBase + t) * 16 + (lane & 15);
        float bs = bias[n];
#pragma unroll
        for (int v = 0; v < 8; ++v) {
            int m = mBlock + mt * 16 + rowBase + v;
            float val = acc[t][v] + bs;
            if constexpr (MODE == 0) {
                outB[(size_t)m * N + n] = (__bf16)val;
            } else if constexpr (MODE == 1) {
                outB[(size_t)m * N + n] = (__bf16)fmaxf(val, 0.0f);
            } else if constexpr (MODE == 2) {
                float wv = ldsEW[mt * 16 + rowBase + v] * fmaxf(val, 0.0f);
                size_t o = (size_t)m * N + n;
                outF[o] = accumulate ? (outF[o] + wv) : wv;
            } else {
                outF[(size_t)m * N + n] = val;
            }
        }
    }
}

// ---------------------------------------------------------------------------
// Row LayerNorm over D=1024, one 256-thread block per row.
// ---------------------------------------------------------------------------
__global__ __launch_bounds__(256) void ln_kernel(
    const float* __restrict__ x, const float* __restrict__ gamma,
    const float* __restrict__ beta, float* __restrict__ out) {
    __shared__ float red[256];
    const int row = blockIdx.x;
    const int tid = threadIdx.x;
    float4 v = ((const float4*)(x + (size_t)row * DIM))[tid];

    red[tid] = v.x + v.y + v.z + v.w;
    __syncthreads();
    for (int o = 128; o > 0; o >>= 1) {
        if (tid < o) red[tid] += red[tid + o];
        __syncthreads();
    }
    float mu = red[0] * (1.0f / DIM);
    __syncthreads();

    float dx = v.x - mu, dy = v.y - mu, dz = v.z - mu, dw = v.w - mu;
    red[tid] = dx * dx + dy * dy + dz * dz + dw * dw;
    __syncthreads();
    for (int o = 128; o > 0; o >>= 1) {
        if (tid < o) red[tid] += red[tid + o];
        __syncthreads();
    }
    float rstd = rsqrtf(red[0] * (1.0f / DIM) + 1e-5f);

    float4 g = ((const float4*)gamma)[tid];
    float4 b = ((const float4*)beta)[tid];
    float4 o;
    o.x = dx * rstd * g.x + b.x;
    o.y = dy * rstd * g.y + b.y;
    o.z = dz * rstd * g.z + b.z;
    o.w = dw * rstd * g.w + b.w;
    ((float4*)(out + (size_t)row * DIM))[tid] = o;
}

// ---------------------------------------------------------------------------
// Host-side orchestration.
// ---------------------------------------------------------------------------
extern "C" void kernel_launch(void* const* d_in, const int* in_sizes, int n_in,
                              void* d_out, int out_size, void* d_ws, size_t ws_size,
                              hipStream_t stream) {
    const float* x     = (const float*)d_in[0];   // [8192,1024]
    const float* ew    = (const float*)d_in[1];   // [8192,8]
    const float* Wi    = (const float*)d_in[2];   // [1024,2048]
    const float* bi    = (const float*)d_in[3];   // [2048]
    const float* W1    = (const float*)d_in[4];   // [8,2048,2048]
    const float* b1    = (const float*)d_in[5];   // [8,2048]
    const float* W2    = (const float*)d_in[6];   // [8,2048,2048]
    const float* b2    = (const float*)d_in[7];   // [8,2048]
    const float* Wo    = (const float*)d_in[8];   // [2048,1024]
    const float* bo    = (const float*)d_in[9];   // [1024]
    const float* gamma = (const float*)d_in[10];  // [1024]
    const float* beta  = (const float*)d_in[11];  // [1024]

    char* wsb = (char*)d_ws;
    size_t off = 0;
    __bf16* hbf   = (__bf16*)(wsb + off); off += (size_t)TOK * HID * 2;
    __bf16* e1bf  = (__bf16*)(wsb + off); off += (size_t)TOK * HID * 2;
    float*  comb  = (float*) (wsb + off); off += (size_t)TOK * HID * 4;
    float*  preLN = (float*) (wsb + off); off += (size_t)TOK * DIM * 4;
    unsigned int* WiP = (unsigned int*)(wsb + off); off += (size_t)DIM * HID * 2;
    unsigned int* W1P = (unsigned int*)(wsb + off); off += (size_t)NE * HID * HID * 2;
    unsigned int* W2P = (unsigned int*)(wsb + off); off += (size_t)NE * HID * HID * 2;
    unsigned int* WoP = (unsigned int*)(wsb + off); off += (size_t)HID * DIM * 2;

    // ---- 1. pack weights (f32 -> bf16 B-fragment layout) ----
    {
        int tWi = DIM * HID / 2;   // dwords
        pack_b_kernel<<<(tWi + 255) / 256, 256, 0, stream>>>(Wi, WiP, HID, 5, tWi);
        int tWo = HID * DIM / 2;
        pack_b_kernel<<<(tWo + 255) / 256, 256, 0, stream>>>(Wo, WoP, DIM, 6, tWo);
        int tE = HID * HID / 2;
        for (int e = 0; e < NE; ++e) {
            pack_b_kernel<<<(tE + 255) / 256, 256, 0, stream>>>(
                W1 + (size_t)e * HID * HID, W1P + (size_t)e * tE, HID, 6, tE);
            pack_b_kernel<<<(tE + 255) / 256, 256, 0, stream>>>(
                W2 + (size_t)e * HID * HID, W2P + (size_t)e * tE, HID, 6, tE);
        }
    }

    // ---- 2. h = x @ Wi + bi  (f32 A, bf16 out) ----
    gemm_bf16_kernel<float, 0, DIM, HID>
        <<<dim3(TOK / 64, HID / 128), 256, 0, stream>>>(
            x, WiP, bi, nullptr, hbf, nullptr, 0, 0);

    // ---- 3. experts: e1 = relu(h@W1[e]+b1), comb (+)= ew*relu(e1@W2[e]+b2) ----
    const size_t eDW = (size_t)HID * HID / 2;
    for (int e = 0; e < NE; ++e) {
        gemm_bf16_kernel<__bf16, 1, HID, HID>
            <<<dim3(TOK / 64, HID / 128), 256, 0, stream>>>(
                hbf, W1P + e * eDW, b1 + (size_t)e * HID, nullptr, e1bf,
                nullptr, e, 0);
        gemm_bf16_kernel<__bf16, 2, HID, HID>
            <<<dim3(TOK / 64, HID / 128), 256, 0, stream>>>(
                e1bf, W2P + e * eDW, b2 + (size_t)e * HID, comb, nullptr, ew,
                e, e > 0 ? 1 : 0);
    }

    // ---- 4. preLN = comb @ Wo + bo (f32 A, f32 out) ----
    gemm_bf16_kernel<float, 3, HID, DIM>
        <<<dim3(TOK / 64, DIM / 128), 256, 0, stream>>>(
            comb, WoP, bo, preLN, nullptr, nullptr, 0, 0);

    // ---- 5. LayerNorm -> d_out ----
    ln_kernel<<<TOK, 256, 0, stream>>>(preLN, gamma, beta, (float*)d_out);
}
